// TrellisLinear_90220083020335
// MI455X (gfx1250) — matrix operand order repaired and checked by reading, not compiled, not run on hardware
//
#include <hip/hip_runtime.h>

typedef __attribute__((ext_vector_type(16))) _Float16 v16h;
typedef __attribute__((ext_vector_type(8)))  _Float16 v8h;
typedef __attribute__((ext_vector_type(4)))  _Float16 v4h;
typedef __attribute__((ext_vector_type(2)))  _Float16 v2h;
typedef __attribute__((ext_vector_type(8)))  float    v8f;

#define DIM   4096
#define BM    128
#define BN    256
#define BK    32
#define PITCH 40                      // BK + 8 f16 pad -> 80B row pitch
#define A_BUF_BYTES (BM * PITCH * 2)  // 10240 bytes per A buffer
#define B_BUF_BYTES (BN * PITCH * 2)  // 20480 bytes per B buffer
#define ROW64_BYTES (64 * PITCH * 2)  // 5120 bytes per 64-row chunk

// ---------------------------------------------------------------------------
// x (fp32) -> xh (f16), 4 elements / thread
// ---------------------------------------------------------------------------
__global__ __launch_bounds__(256) void convert_x_kernel(const float* __restrict__ x,
                                                        _Float16* __restrict__ xh) {
  size_t i = ((size_t)blockIdx.x * 256 + threadIdx.x) * 4;
  float4 v = *(const float4*)(x + i);
  v4h h = { (_Float16)v.x, (_Float16)v.y, (_Float16)v.z, (_Float16)v.w };
  *(v4h*)(xh + i) = h;
}

// ---------------------------------------------------------------------------
// Dequantize packed 4-bit indices -> Wh[k][n] f16 row-major.
// W[k][n] = grid[idx] * scales[n>>7][k] * su[n] * sv[k]
// ---------------------------------------------------------------------------
__global__ __launch_bounds__(256) void dequant_kernel(const int*   __restrict__ packed,
                                                      const float* __restrict__ scales,
                                                      const float* __restrict__ su,
                                                      const float* __restrict__ sv,
                                                      const float* __restrict__ cb,
                                                      _Float16*    __restrict__ Wh) {
  int t  = blockIdx.x * 256 + threadIdx.x;   // 0 .. 256*256*128-1
  int j  = t & 127;
  int tn = (t >> 7) & 255;
  int tk = t >> 15;
  int v  = packed[t];
  int o0 = j << 1;            // even local offset
  int lk = o0 >> 4;
  int ln = o0 & 15;
  int k  = (tk << 4) + lk;
  int n  = (tn << 4) + ln;
  float sc = scales[(n >> 7) * DIM + k] * sv[k];
  float w0 = cb[v & 15]        * sc * su[n];
  float w1 = cb[(v >> 4) & 15] * sc * su[n + 1];
  v2h h = { (_Float16)w0, (_Float16)w1 };
  *(v2h*)(Wh + (size_t)k * DIM + n) = h;
}

// ---------------------------------------------------------------------------
// Helpers: LDS byte offset of a __shared__ element; async global->LDS b128.
// ---------------------------------------------------------------------------
__device__ __forceinline__ unsigned ldsOffset(const void* p) {
  return (unsigned)(unsigned long long)
         (__attribute__((address_space(3))) const void*)p;
}

__device__ __forceinline__ void asyncLoad16(unsigned ldsDst, const _Float16* gsrc) {
  // GLOBAL_LOAD_ASYNC_TO_LDS_B128: per-lane 16B memory -> LDS, ASYNCcnt-tracked
  asm volatile("global_load_async_to_lds_b128 %0, %1, off"
               :: "v"(ldsDst), "v"((unsigned long long)(uintptr_t)gsrc)
               : "memory");
}

__device__ __forceinline__ void waitAsync0() {
  asm volatile("s_wait_asynccnt 0" ::: "memory");
}

// ---------------------------------------------------------------------------
// C[m][o] = sum_n xh[m][n] * Wh[o][n]  (f16 in, f32 accum via WMMA)
// 256 threads = 8 waves; block tile 128x256; wave tile 64x64 (4x4 fragments)
// -> 16 WMMA per 16 ds_load_b128 per k-step. Double-buffered LDS fed by
// async global->LDS b128 (ASYNCcnt).
// ---------------------------------------------------------------------------
__global__ __launch_bounds__(256) void gemm_kernel(const _Float16* __restrict__ A,
                                                   const _Float16* __restrict__ B,
                                                   float* __restrict__ C) {
  __shared__ _Float16 sA[2][BM * PITCH];   // 20 KB
  __shared__ _Float16 sB[2][BN * PITCH];   // 40 KB

  const int tid  = threadIdx.x;
  const int lane = tid & 31;
  const int wv   = tid >> 5;   // 0..7
  const int wm   = wv >> 2;    // 0..1  (64 rows each)
  const int wn   = wv & 3;     // 0..3  (64 cols each)
  const int lm   = lane & 15;
  const int lh   = lane >> 4;

  const int mBase = blockIdx.y * BM;
  const int nBase = blockIdx.x * BN;

  // cooperative global->LDS mapping: per thread, 16B chunks at
  // row = tid>>2 (+64*i), seg = (tid&3)*8 f16 within BK
  const int row = tid >> 2;            // 0..63
  const int seg = (tid & 3) * 8;
  const _Float16* aSrc = A + (size_t)(mBase + row) * DIM + seg;
  const _Float16* bSrc = B + (size_t)(nBase + row) * DIM + seg;
  const int ldsOff = row * PITCH + seg;

  const unsigned aDst0 = ldsOffset(&sA[0][ldsOff]);
  const unsigned bDst0 = ldsOffset(&sB[0][ldsOff]);

  v8f zero = {};
  v8f acc[4][4];
#pragma unroll
  for (int i = 0; i < 4; ++i)
#pragma unroll
    for (int jj = 0; jj < 4; ++jj) acc[i][jj] = zero;

  // prologue: k-tile 0 -> buffer 0 (A: 2 chunks, B: 4 chunks per thread)
  asyncLoad16(aDst0,               aSrc);
  asyncLoad16(aDst0 + ROW64_BYTES, aSrc + (size_t)64 * DIM);
#pragma unroll
  for (int i = 0; i < 4; ++i)
    asyncLoad16(bDst0 + i * ROW64_BYTES, bSrc + (size_t)(64 * i) * DIM);
  waitAsync0();
  __syncthreads();

  const int KT = DIM / BK;  // 128
  for (int kt = 0; kt < KT; ++kt) {
    const int cur = kt & 1;
    const int nxt = cur ^ 1;

    // async-prefetch next k-tile directly into the other LDS buffer
    if (kt + 1 < KT) {
      const int kk = (kt + 1) * BK;
      const unsigned aD = aDst0 + (unsigned)nxt * A_BUF_BYTES;
      const unsigned bD = bDst0 + (unsigned)nxt * B_BUF_BYTES;
      asyncLoad16(aD,               aSrc + kk);
      asyncLoad16(aD + ROW64_BYTES, aSrc + (size_t)64 * DIM + kk);
#pragma unroll
      for (int i = 0; i < 4; ++i)
        asyncLoad16(bD + i * ROW64_BYTES, bSrc + (size_t)(64 * i) * DIM + kk);
    }

    // A fragments: lane half h -> K = 16h..16h+15 contiguous
    v16h af[4];
#pragma unroll
    for (int mt = 0; mt < 4; ++mt) {
      const _Float16* p = &sA[cur][(wm * 64 + mt * 16 + lm) * PITCH];
      v8h lo = *(const v8h*)(p + 16 * lh);
      v8h hi = *(const v8h*)(p + 16 * lh + 8);
      af[mt] = __builtin_shufflevector(lo, hi, 0,1,2,3,4,5,6,7,8,9,10,11,12,13,14,15);
    }
    // B fragments: column n = lane%16, K = 16h..16h+15 contiguous
    v16h bf[4];
#pragma unroll
    for (int nt = 0; nt < 4; ++nt) {
      const _Float16* p = &sB[cur][(wn * 64 + nt * 16 + lm) * PITCH + 16 * lh];
      v8h lo = *(const v8h*)(p);
      v8h hi = *(const v8h*)(p + 8);
      bf[nt] = __builtin_shufflevector(lo, hi, 0,1,2,3,4,5,6,7,8,9,10,11,12,13,14,15);
    }

#pragma unroll
    for (int mt = 0; mt < 4; ++mt)
#pragma unroll
      for (int nt = 0; nt < 4; ++nt)
        acc[mt][nt] = __builtin_amdgcn_wmma_f32_16x16x32_f16(
            false, af[mt], false, bf[nt], (short)0, acc[mt][nt], false, false);

    // async writes to nxt must land before anyone reads them next iteration
    waitAsync0();
    __syncthreads();
  }

  // C/D layout: m = vgpr + 8*lh, n = lane%16
#pragma unroll
  for (int mt = 0; mt < 4; ++mt) {
#pragma unroll
    for (int nt = 0; nt < 4; ++nt) {
      const int col = nBase + wn * 64 + nt * 16 + lm;
      const int r0  = mBase + wm * 64 + mt * 16 + lh * 8;
#pragma unroll
      for (int v = 0; v < 8; ++v)
        C[(size_t)(r0 + v) * DIM + col] = acc[mt][nt][v];
    }
  }
}

// ---------------------------------------------------------------------------
extern "C" void kernel_launch(void* const* d_in, const int* in_sizes, int n_in,
                              void* d_out, int out_size, void* d_ws, size_t ws_size,
                              hipStream_t stream) {
  const float* x      = (const float*)d_in[0];
  const int*   packed = (const int*)  d_in[1];
  const float* scales = (const float*)d_in[2];
  const float* su     = (const float*)d_in[3];
  const float* sv     = (const float*)d_in[4];
  const float* grid   = (const float*)d_in[5];
  float*       out    = (float*)d_out;

  _Float16* xh = (_Float16*)d_ws;                       // 32 MB
  _Float16* Wh = xh + (size_t)DIM * DIM;                // 32 MB

  convert_x_kernel<<<(DIM * DIM / 4) / 256, 256, 0, stream>>>(x, xh);
  dequant_kernel<<<(256 * 256 * 128) / 256, 256, 0, stream>>>(packed, scales, su, sv, grid, Wh);

  dim3 g(DIM / BN, DIM / BM);  // (16, 32)
  gemm_kernel<<<g, 256, 0, stream>>>(xh, Wh, out);
}
